// VQVAE_64931315581381
// MI455X (gfx1250) — compile-verified, matrix operand order checked by reading, not run on hardware
//
#include <hip/hip_runtime.h>
#include <hip/hip_bf16.h>
#include <math.h>

// ---------------------------------------------------------------------------
// VQ-VAE forward for MI455X (gfx1250, wave32, WMMA).
// All dense math goes through v_wmma_f32_16x16x32_f16 (f16 in, f32 accum),
// converting f32 operands to f16 while staging tiles into LDS.
// GEMM: block tile 64x64, BK=64; 8 waves (4x2), each wave 16x32 (2 accums).
// LDS double-buffered: global loads of tile t+1 overlap WMMAs of tile t,
// one barrier per k-iteration. B staged n-major so fragments are ds_load_b128.
// ---------------------------------------------------------------------------

#define BB   32
#define TT   512
#define BT   (BB * TT)      // 16384
#define DD   512
#define NHH  8
#define DHH  64
#define NLL  4
#define NQQ  4
#define CBB  1024
#define DATAN 256
#define FFN  2048

typedef __attribute__((ext_vector_type(16))) _Float16 v16h;
typedef __attribute__((ext_vector_type(8)))  float    v8f;

// ---------------------------------------------------------------------------
// Generic batched GEMM:  C[z] = act(A[z] (MxK) * B[z] (KxN) + bias)
// Batch pointer:  zo = z / innerDiv, zi = z % innerDiv
//   base += zo*outerStride + zi*innerStride    (lets heads index qkv directly)
// transB: B[k][n] = Bsrc[n*ldb + k]  (for Q*K^T and r*cb^T)
// act: 0 = none, 1 = relu, 2 = exact gelu
// ---------------------------------------------------------------------------
__global__ __launch_bounds__(256) void gemm_wmma_f16(
    const float* __restrict__ A, const float* __restrict__ Bm,
    const float* __restrict__ bias, float* __restrict__ C,
    int M, int N, int K, int lda, int ldb, int ldc,
    long long aInner, long long bInner, long long cInner, int innerDiv,
    long long aOuter, long long bOuter, long long cOuter,
    int transB, int act)
{
    const int z  = blockIdx.z;
    const int zo = z / innerDiv;
    const int zi = z - zo * innerDiv;
    const float* Ab = A  + (long long)zo * aOuter + (long long)zi * aInner;
    const float* Bb = Bm + (long long)zo * bOuter + (long long)zi * bInner;
    float*       Cb = C  + (long long)zo * cOuter + (long long)zi * cInner;

    const int mBase = blockIdx.x * 64;
    const int nBase = blockIdx.y * 64;

    __shared__ _Float16 As[2][64][72];   // m-major: 64 rows x 64 K (+8 pad)
    __shared__ _Float16 Bs[2][64][72];   // n-major: 64 cols x 64 K (+8 pad)

    const int tid  = threadIdx.x;
    const int lane = tid & 31;
    const int wid  = tid >> 5;
    const int wm   = wid >> 1;         // 0..3  (M tile of 16)
    const int wn   = wid & 1;          // 0..1  (N tile of 32 = 2 sub-tiles)
    const int fr   = lane & 15;        // fragment row/col within 16
    const int kb   = (lane >> 4) << 3; // 0 or 8 (K sub-block per WMMA layout)

    v8f acc0 = {0.f, 0.f, 0.f, 0.f, 0.f, 0.f, 0.f, 0.f};
    v8f acc1 = {0.f, 0.f, 0.f, 0.f, 0.f, 0.f, 0.f, 0.f};

    // A loader: thread -> row tid/4 (0..63), 16 consecutive K
    const int ar = tid >> 2;
    const int ac = (tid & 3) << 4;
    // B loader: thread -> col n = tid&63, K chunk of 16
    const int bn = tid & 63;
    const int bk = (tid >> 6) << 4;

    float ra[16], rb[16];   // register staging for the in-flight tile

    auto loadRegs = [&](int k0) {
        const float* asrc = Ab + (long long)(mBase + ar) * lda + (k0 + ac);
        if (k0 + 64 < K) __builtin_prefetch(asrc + 64, 0, 1);  // global_prefetch_b8
        #pragma unroll
        for (int j = 0; j < 16; ++j) ra[j] = asrc[j];
        if (!transB) {
            // row-major B: per j, lanes with consecutive n hit consecutive addrs
            #pragma unroll
            for (int j = 0; j < 16; ++j)
                rb[j] = Bb[(long long)(k0 + bk + j) * ldb + (nBase + bn)];
        } else {
            const float* bsrc = Bb + (long long)(nBase + bn) * ldb + (k0 + bk);
            #pragma unroll
            for (int j = 0; j < 16; ++j) rb[j] = bsrc[j];
        }
    };
    auto storeTile = [&](int buf) {
        #pragma unroll
        for (int j = 0; j < 16; ++j) As[buf][ar][ac + j] = (_Float16)ra[j];
        #pragma unroll
        for (int j = 0; j < 16; ++j) Bs[buf][bn][bk + j] = (_Float16)rb[j];
    };
    auto compute = [&](int buf) {
        #pragma unroll
        for (int s = 0; s < 2; ++s) {
            const int ko = s * 32;
            v16h a, b0, b1;
            #pragma unroll
            for (int j = 0; j < 8; ++j) {
                a[j]      = As[buf][wm * 16 + fr][ko + kb + j];
                a[8 + j]  = As[buf][wm * 16 + fr][ko + 16 + kb + j];
                b0[j]     = Bs[buf][wn * 32 + fr][ko + kb + j];
                b0[8 + j] = Bs[buf][wn * 32 + fr][ko + 16 + kb + j];
                b1[j]     = Bs[buf][wn * 32 + 16 + fr][ko + kb + j];
                b1[8 + j] = Bs[buf][wn * 32 + 16 + fr][ko + 16 + kb + j];
            }
            acc0 = __builtin_amdgcn_wmma_f32_16x16x32_f16(
                     false, a, false, b0, (short)0, acc0, false, false);
            acc1 = __builtin_amdgcn_wmma_f32_16x16x32_f16(
                     false, a, false, b1, (short)0, acc1, false, false);
        }
    };

    // software pipeline: tile t+1 global loads overlap tile t WMMAs
    const int nt = K >> 6;            // K is always a multiple of 64 here
    loadRegs(0);
    storeTile(0);
    for (int t = 0; t < nt; ++t) {
        __syncthreads();              // tile t visible in LDS buf (t&1)
        const int cur = t & 1;
        if (t + 1 < nt) loadRegs((t + 1) << 6);   // uniform guard
        compute(cur);
        if (t + 1 < nt) storeTile(cur ^ 1);
    }

    // epilogue: C/D layout -> lane 0-15: N=lane, M=r ; lane 16-31: N=lane-16, M=8+r
    const int m0 = mBase + wm * 16 + ((lane >> 4) << 3);
    #pragma unroll
    for (int sub = 0; sub < 2; ++sub) {
        const int cn = nBase + wn * 32 + sub * 16 + (lane & 15);
        const float bv = bias ? bias[cn] : 0.f;
        const v8f& acc = sub ? acc1 : acc0;
        #pragma unroll
        for (int r = 0; r < 8; ++r) {
            float v = acc[r] + bv;
            if (act == 1)      v = fmaxf(v, 0.f);
            else if (act == 2) v = 0.5f * v * (1.f + erff(v * 0.70710678118f));
            Cb[(long long)(m0 + r) * ldc + cn] = v;
        }
    }
}

// ---------------------------------------------------------------------------
// Sinusoidal positional embedding add (computed on the fly).
// ---------------------------------------------------------------------------
__global__ void pos_add_k(float* __restrict__ h)
{
    const long long i = (long long)blockIdx.x * 256 + threadIdx.x;  // BT*D elems
    const int d = (int)(i & 511);
    const int t = (int)((i >> 9) & 511);
    const int pair = d >> 1;
    const float div = expf((float)(2 * pair) * (-9.2103403719761836f / 512.0f));
    const float ang = (float)t * div;
    h[i] += (d & 1) ? cosf(ang) : sinf(ang);
}

// ---------------------------------------------------------------------------
// Causal softmax over one score row (in place, f32). row = head*T + q.
// ---------------------------------------------------------------------------
__global__ void softmax_causal_k(float* __restrict__ s, float scale)
{
    const int row = blockIdx.x;
    const int q   = row & (TT - 1);
    float* p = s + (long long)row * TT;
    const int t = threadIdx.x;
    __shared__ float red[256];

    float v0 = (t       <= q) ? p[t]       * scale : -3.0e38f;
    float v1 = (t + 256 <= q) ? p[t + 256] * scale : -3.0e38f;

    red[t] = fmaxf(v0, v1); __syncthreads();
    for (int st = 128; st > 0; st >>= 1) { if (t < st) red[t] = fmaxf(red[t], red[t + st]); __syncthreads(); }
    const float rmax = red[0]; __syncthreads();

    float e0 = (t       <= q) ? expf(v0 - rmax) : 0.f;
    float e1 = (t + 256 <= q) ? expf(v1 - rmax) : 0.f;
    red[t] = e0 + e1; __syncthreads();
    for (int st = 128; st > 0; st >>= 1) { if (t < st) red[t] += red[t + st]; __syncthreads(); }
    const float inv = 1.f / red[0];

    p[t]       = e0 * inv;
    p[t + 256] = e1 * inv;
}

// ---------------------------------------------------------------------------
// LayerNorm over D=512 per row; optional residual add. dst may alias src.
// ---------------------------------------------------------------------------
__global__ void row_ln_k(const float* __restrict__ src, const float* __restrict__ add,
                         float* __restrict__ dst,
                         const float* __restrict__ g, const float* __restrict__ b)
{
    const int row = blockIdx.x;
    const int t   = threadIdx.x;
    const float* s = src + (long long)row * DD;
    float x0 = s[t], x1 = s[t + 256];
    if (add) { const float* a = add + (long long)row * DD; x0 += a[t]; x1 += a[t + 256]; }

    __shared__ float red[256];
    red[t] = x0 + x1; __syncthreads();
    for (int st = 128; st > 0; st >>= 1) { if (t < st) red[t] += red[t + st]; __syncthreads(); }
    const float mean = red[0] * (1.f / (float)DD); __syncthreads();

    const float d0 = x0 - mean, d1 = x1 - mean;
    red[t] = d0 * d0 + d1 * d1; __syncthreads();
    for (int st = 128; st > 0; st >>= 1) { if (t < st) red[t] += red[t + st]; __syncthreads(); }
    const float rstd = rsqrtf(red[0] * (1.f / (float)DD) + 1e-5f);

    float* dp = dst + (long long)row * DD;
    dp[t]       = d0 * rstd * g[t]       + b[t];
    dp[t + 256] = d1 * rstd * g[t + 256] + b[t + 256];
}

// ---------------------------------------------------------------------------
// Codebook squared norms: one row (512 elems) per block. Covers NQ*CB rows.
// ---------------------------------------------------------------------------
__global__ void cbnorm_k(const float* __restrict__ cb, float* __restrict__ norms)
{
    const int row = blockIdx.x;
    const int t   = threadIdx.x;
    const float* p = cb + (long long)row * DD;
    __shared__ float red[256];
    red[t] = p[t] * p[t] + p[t + 256] * p[t + 256]; __syncthreads();
    for (int st = 128; st > 0; st >>= 1) { if (t < st) red[t] += red[t + st]; __syncthreads(); }
    if (t == 0) norms[row] = red[0];
}

// ---------------------------------------------------------------------------
// Per-row argmin of ||cb_n||^2 - 2*dot(r, cb_n)  over CB=1024.
// ---------------------------------------------------------------------------
__global__ void argmin_k(const float* __restrict__ dots, const float* __restrict__ norms,
                         int* __restrict__ idxbuf, float* __restrict__ outIdx, int qi)
{
    const int row = blockIdx.x;
    const int t   = threadIdx.x;
    const float* dp = dots + (long long)row * CBB;
    const float* np = norms + qi * CBB;
    float best = 3.0e38f; int bi = 0;
    for (int j = t; j < CBB; j += 256) {
        const float v = np[j] - 2.f * dp[j];
        if (v < best) { best = v; bi = j; }
    }
    __shared__ float bv[256]; __shared__ int bix[256];
    bv[t] = best; bix[t] = bi; __syncthreads();
    for (int st = 128; st > 0; st >>= 1) {
        if (t < st) {
            if (bv[t + st] < bv[t] || (bv[t + st] == bv[t] && bix[t + st] < bix[t])) {
                bv[t] = bv[t + st]; bix[t] = bix[t + st];
            }
        }
        __syncthreads();
    }
    if (t == 0) {
        idxbuf[row] = bix[0];
        outIdx[(long long)row * NQQ + qi] = (float)bix[0];
    }
}

// ---------------------------------------------------------------------------
// Rotation trick update for one quantizer step:
//   quant = rotate_to(residual, cb[idx]);  residual -= quant;  qout += quant
// Also accumulates commitment loss Sum((qv - residual)^2) into lossAcc[qi].
// ---------------------------------------------------------------------------
__global__ void rotate_k(float* __restrict__ residual, float* __restrict__ qout,
                         const float* __restrict__ cb, const int* __restrict__ idxbuf,
                         float* __restrict__ lossAcc, int qi, int first)
{
    const int row = blockIdx.x;
    const int t   = threadIdx.x;
    float* x = residual + (long long)row * DD;
    const float* q = cb + (long long)idxbuf[row] * DD;
    const float x0 = x[t], x1 = x[t + 256];
    const float q0 = q[t], q1 = q[t + 256];

    __shared__ float r0[256], r1[256], r2[256], r3[256];
    const float e0 = q0 - x0, e1 = q1 - x1;
    r0[t] = x0 * x0 + x1 * x1;
    r1[t] = q0 * q0 + q1 * q1;
    r2[t] = x0 * q0 + x1 * q1;
    r3[t] = e0 * e0 + e1 * e1;
    __syncthreads();
    for (int st = 128; st > 0; st >>= 1) {
        if (t < st) { r0[t] += r0[t+st]; r1[t] += r1[t+st]; r2[t] += r2[t+st]; r3[t] += r3[t+st]; }
        __syncthreads();
    }
    const float sxx = r0[0], sqq = r1[0], sxq = r2[0], sdiff = r3[0];

    const float eps = 1e-6f;
    const float nx  = sqrtf(sxx), nq = sqrtf(sqq);
    const float nxc = fmaxf(nx, eps), nqc = fmaxf(nq, eps);
    const float ww  = sxx / (nxc * nxc) + 2.f * sxq / (nxc * nqc) + sqq / (nqc * nqc);
    const float nw  = fmaxf(sqrtf(ww), eps);
    const float xw  = (sxx / nxc + sxq / nqc) / nw;   // x . w_hat
    const float xu  = sxx / nxc;                       // x . u
    const float sc  = nq / nxc;

    const float w0 = (x0 / nxc + q0 / nqc) / nw;
    const float w1 = (x1 / nxc + q1 / nqc) / nw;
    const float o0 = (x0 - 2.f * xw * w0 + 2.f * xu * (q0 / nqc)) * sc;
    const float o1 = (x1 - 2.f * xw * w1 + 2.f * xu * (q1 / nqc)) * sc;

    if (t == 0) atomicAdd(&lossAcc[qi], sdiff);

    x[t]       = x0 - o0;
    x[t + 256] = x1 - o1;
    float* qo = qout + (long long)row * DD;
    if (first) { qo[t] = o0; qo[t + 256] = o1; }
    else       { qo[t] += o0; qo[t + 256] += o1; }
}

__global__ void zero_k(float* __restrict__ p, int n)
{
    const int i = blockIdx.x * 256 + threadIdx.x;
    if (i < n) p[i] = 0.f;
}

__global__ void finalize_commit_k(const float* __restrict__ lossAcc, float* __restrict__ out)
{
    if (threadIdx.x == 0 && blockIdx.x == 0) {
        float s = 0.f;
        #pragma unroll
        for (int i = 0; i < NQQ; ++i)
            s += 0.25f * lossAcc[i] / (float)((long long)BT * DD);
        out[0] = s / (float)NQQ;   // mean over quantizers
    }
}

// ---------------------------------------------------------------------------
// Host orchestration
// ---------------------------------------------------------------------------
extern "C" void kernel_launch(void* const* d_in, const int* in_sizes, int n_in,
                              void* d_out, int out_size, void* d_ws, size_t ws_size,
                              hipStream_t stream)
{
    const float* x        = (const float*)d_in[0];
    const float* w_in     = (const float*)d_in[1];
    const float* b_in     = (const float*)d_in[2];
    const float* qkv_w    = (const float*)d_in[3];
    const float* qkv_b    = (const float*)d_in[4];
    const float* out_w    = (const float*)d_in[5];
    const float* out_b    = (const float*)d_in[6];
    const float* ln1_s    = (const float*)d_in[7];
    const float* ln1_b    = (const float*)d_in[8];
    const float* ln2_s    = (const float*)d_in[9];
    const float* ln2_b    = (const float*)d_in[10];
    const float* ff1_w    = (const float*)d_in[11];
    const float* ff1_b    = (const float*)d_in[12];
    const float* ff2_w    = (const float*)d_in[13];
    const float* ff2_b    = (const float*)d_in[14];
    const float* codebooks= (const float*)d_in[15];
    const float* dec_ln_s = (const float*)d_in[16];
    const float* dec_ln_b = (const float*)d_in[17];
    const float* dec_w1   = (const float*)d_in[18];
    const float* dec_b1   = (const float*)d_in[19];
    const float* dec_w2   = (const float*)d_in[20];
    const float* dec_b2   = (const float*)d_in[21];
    float* out = (float*)d_out;

    // ---- workspace layout (bump allocated, ~244 MB peak) ----
    char* wsb = (char*)d_ws;
    size_t off = 0;
    auto alloc = [&](size_t bytes) -> char* {
        char* p = wsb + off;
        off += (bytes + 255) & ~(size_t)255;
        return p;
    };
    float* h       = (float*)alloc((size_t)BT * DD * 4);  // hidden; later VQ residual (in place)
    float* tmp     = (float*)alloc((size_t)BT * DD * 4);  // GEMM output pre-LN; dec hidden
    float* qout    = (float*)alloc((size_t)BT * DD * 4);  // quant_out accumulator
    int*   idxbuf  = (int*)  alloc((size_t)BT * 4);
    float* lossAcc = (float*)alloc(256);
    float* norms   = (float*)alloc((size_t)NQQ * CBB * 4);
    // big region, reused per phase:
    //   attention: qkv (BT x 1536 f32) | scores (8 x T x T f32) | o (BT x D f32)
    //   ffn:       ffh (BT x 2048 f32)
    //   vq:        dots (BT x CB f32)
    //   decoder:   dec1 (BT x D f32)
    char* big = alloc((size_t)BT * 1536 * 4 + (size_t)NHH * TT * TT * 4 + (size_t)BT * DD * 4);
    float* qkv    = (float*)big;
    float* scores = (float*)(big + (size_t)BT * 1536 * 4);
    float* o      = (float*)(big + (size_t)BT * 1536 * 4 + (size_t)NHH * TT * TT * 4);
    float* ffh    = (float*)big;
    float* dots   = (float*)big;
    float* dec1   = (float*)big;

    auto gemm = [&](const float* A, const float* Bm, const float* bias, float* C,
                    int M, int N, int K, int lda, int ldb, int ldc,
                    int batches, long long aIn, long long bIn, long long cIn, int innerDiv,
                    int transB, int act) {
        dim3 grid(M / 64, N / 64, batches);
        gemm_wmma_f16<<<grid, dim3(256), 0, stream>>>(
            A, Bm, bias, C, M, N, K, lda, ldb, ldc,
            aIn, bIn, cIn, innerDiv, 0LL, 0LL, 0LL, transB, act);
    };

    // ---- input projection + positional embedding ----
    gemm(x, w_in, b_in, h, BT, DD, DATAN, DATAN, DD, DD, 1, 0, 0, 0, 1, 0, 0);
    pos_add_k<<<(BT * DD) / 256, 256, 0, stream>>>(h);

    // ---- transformer layers ----
    for (int l = 0; l < NLL; ++l) {
        gemm(h, qkv_w + (size_t)l * DD * 3 * DD, qkv_b + (size_t)l * 3 * DD, qkv,
             BT, 3 * DD, DD, DD, 3 * DD, 3 * DD, 1, 0, 0, 0, 1, 0, 0);

        // attention, chunked over batch b (keeps scores buffer to 8 heads)
        for (int b = 0; b < BB; ++b) {
            const float* qb = qkv + (size_t)b * TT * 3 * DD;
            // S = Q K^T  : per-head z, A/B offset by z*64 inside the qkv row
            gemm(qb, qb + DD, nullptr, scores,
                 TT, TT, DHH, 3 * DD, 3 * DD, TT,
                 NHH, 64LL, 64LL, (long long)TT * TT, NHH, 1, 0);
            softmax_causal_k<<<NHH * TT, 256, 0, stream>>>(scores, 0.125f);
            // O = P V : writes directly into merged [T, D] layout for this b
            gemm(scores, qb + 2 * DD, nullptr, o + (size_t)b * TT * DD,
                 TT, DHH, TT, TT, 3 * DD, DD,
                 NHH, (long long)TT * TT, 64LL, 64LL, NHH, 0, 0);
        }

        gemm(o, out_w + (size_t)l * DD * DD, out_b + (size_t)l * DD, tmp,
             BT, DD, DD, DD, DD, DD, 1, 0, 0, 0, 1, 0, 0);
        row_ln_k<<<BT, 256, 0, stream>>>(h, tmp, h, ln1_s + l * DD, ln1_b + l * DD);

        gemm(h, ff1_w + (size_t)l * DD * FFN, ff1_b + (size_t)l * FFN, ffh,
             BT, FFN, DD, DD, FFN, FFN, 1, 0, 0, 0, 1, 0, 1 /*relu*/);
        gemm(ffh, ff2_w + (size_t)l * FFN * DD, ff2_b + (size_t)l * DD, tmp,
             BT, DD, FFN, FFN, DD, DD, 1, 0, 0, 0, 1, 0, 0);
        row_ln_k<<<BT, 256, 0, stream>>>(h, tmp, h, ln2_s + l * DD, ln2_b + l * DD);
    }

    // ---- residual VQ with rotation trick ----
    zero_k<<<1, 256, 0, stream>>>(lossAcc, NQQ);
    cbnorm_k<<<NQQ * CBB, 256, 0, stream>>>(codebooks, norms);
    for (int qi = 0; qi < NQQ; ++qi) {
        const float* cbq = codebooks + (size_t)qi * CBB * DD;
        // dot(r, cb^T): [BT, D] x [D, CB] with transB
        gemm(h, cbq, nullptr, dots, BT, CBB, DD, DD, DD, CBB, 1, 0, 0, 0, 1, 1, 0);
        argmin_k<<<BT, 256, 0, stream>>>(dots, norms, idxbuf,
                                         out + (size_t)BT * DATAN + 1, qi);
        rotate_k<<<BT, 256, 0, stream>>>(h, qout, cbq, idxbuf, lossAcc, qi,
                                         (qi == 0) ? 1 : 0);
    }
    finalize_commit_k<<<1, 32, 0, stream>>>(lossAcc, out + (size_t)BT * DATAN);

    // ---- decoder ----
    row_ln_k<<<BT, 256, 0, stream>>>(qout, nullptr, tmp, dec_ln_s, dec_ln_b);
    gemm(tmp, dec_w1, dec_b1, dec1, BT, DD, DD, DD, DD, DD, 1, 0, 0, 0, 1, 0, 2 /*gelu*/);
    gemm(dec1, dec_w2, dec_b2, out, BT, DATAN, DD, DD, DATAN, DATAN, 1, 0, 0, 0, 1, 0, 0);
}
